// TaskAttention_523986010403
// MI455X (gfx1250) — matrix operand order repaired and checked by reading, not compiled, not run on hardware
//
#include <hip/hip_runtime.h>

// ---------------------------------------------------------------------------
// CDNA5 (gfx1250) wave32 WMMA implementation of task-masked MHA + LN.
// All GEMMs use v_wmma_f32_16x16x32_bf16 (bf16 A/B, f32 accumulate).
// Attention K/V tiles are double-buffered via GLOBAL_LOAD_ASYNC_TO_LDS_B128
// (ASYNCcnt) when the toolchain exposes the builtins.
// ---------------------------------------------------------------------------

typedef __attribute__((ext_vector_type(16))) __bf16 bf16x16;
typedef __attribute__((ext_vector_type(8)))  float  floatx8;

union ABf { unsigned u[8]; bf16x16 v; };

#define WMMA_BF16(A, B, C) \
  __builtin_amdgcn_wmma_f32_16x16x32_bf16(false, (A), false, (B), (short)0, (C), false, false)

#if defined(__has_builtin)
#  if __has_builtin(__builtin_amdgcn_global_load_async_to_lds_b128) && \
      __has_builtin(__builtin_amdgcn_s_wait_asynccnt)
#    define HAVE_ASYNC_LDS 1
#  endif
#endif
#ifndef HAVE_ASYNC_LDS
#  define HAVE_ASYNC_LDS 0
#endif

#if HAVE_ASYNC_LDS
typedef int v4i_ __attribute__((vector_size(16)));
#define GPTR(p) ((__attribute__((address_space(1))) v4i_*)(p))
#define LPTR(p) ((__attribute__((address_space(3))) v4i_*)(p))
#define ASYNC_WAIT() __builtin_amdgcn_s_wait_asynccnt(0)
#else
#define ASYNC_WAIT()
#endif

__device__ __forceinline__ unsigned short f2bf(float f) {
  unsigned u = __float_as_uint(f);
  unsigned r = 0x7FFFu + ((u >> 16) & 1u);   // round-to-nearest-even
  return (unsigned short)((u + r) >> 16);
}
__device__ __forceinline__ unsigned pack2bf(float a, float b) {
  return (unsigned)f2bf(a) | ((unsigned)f2bf(b) << 16);
}

// Problem constants
#define S_TOT 2048
#define BATCH 8
#define DMODEL 1024
#define NHEAD 16
#define HDIM 64
#define SEP 1024
#define NTOK (S_TOT * BATCH)        // 16384
#define NKV  (SEP * BATCH)          // 8192

#define ASTR 40                      // LDS row stride (elems), 80B: 16B aligned
#define KSTR 72                      // LDS row stride (elems), 144B: 16B aligned

// ---------------------------------------------------------------------------
// Kernel 1: QKV projection.  C[t][o] = sum_d src[t][d] * W[o][d] + bias[o]
// M = 16384 tokens, N = 3072 (Q|K|V), K = 1024.  128x128 block tiles,
// 8 waves each computing 32x64 via 2x4 WMMA accumulators.
// ---------------------------------------------------------------------------
__global__ __launch_bounds__(256) void qkv_gemm_kernel(
    const float* __restrict__ src, const float* __restrict__ w,
    const float* __restrict__ bias,
    unsigned short* __restrict__ qb,   // (B,H,2048,64) bf16
    unsigned short* __restrict__ kb,   // (B,H,1024,64) bf16
    unsigned short* __restrict__ vb)   // (B,H,64,1024) bf16 (transposed)
{
  const int n0 = blockIdx.x * 128;
  const int m0 = blockIdx.y * 128;
  if (n0 >= 1024 && m0 >= NKV) return;   // K/V only needed for train tokens

  __shared__ __align__(16) unsigned short As[128 * ASTR];
  __shared__ __align__(16) unsigned short Bs[128 * ASTR];

  const int tid = threadIdx.x;
  const int wv = tid >> 5, lane = tid & 31, hi = lane >> 4, ln = lane & 15;
  const int wm = wv >> 1, wn = wv & 1;

  floatx8 zero = {};
  floatx8 acc[2][4];
  for (int i = 0; i < 2; ++i)
    for (int j = 0; j < 4; ++j) acc[i][j] = zero;

  const int lrow = tid >> 1;
  const int lcg  = (tid & 1) * 16;

  for (int kt = 0; kt < DMODEL; kt += 32) {
    __syncthreads();
    {
      const float* p = src + (size_t)(m0 + lrow) * DMODEL + kt + lcg;
      const float* q = w   + (size_t)(n0 + lrow) * DMODEL + kt + lcg;
#pragma unroll
      for (int j = 0; j < 4; ++j) {
        float4 f = *(const float4*)(p + j * 4);
        *(unsigned*)(&As[lrow * ASTR + lcg + j * 4    ]) = pack2bf(f.x, f.y);
        *(unsigned*)(&As[lrow * ASTR + lcg + j * 4 + 2]) = pack2bf(f.z, f.w);
        float4 g = *(const float4*)(q + j * 4);
        *(unsigned*)(&Bs[lrow * ASTR + lcg + j * 4    ]) = pack2bf(g.x, g.y);
        *(unsigned*)(&Bs[lrow * ASTR + lcg + j * 4 + 2]) = pack2bf(g.z, g.w);
      }
      if (kt + 32 < DMODEL) {             // hint next K-step tiles into cache
        __builtin_prefetch(p + 32, 0, 0);
        __builtin_prefetch(q + 32, 0, 0);
      }
    }
    __syncthreads();

    ABf a[2], b[4];
#pragma unroll
    for (int mt = 0; mt < 2; ++mt) {
      const int r = wm * 32 + mt * 16 + ln;
#pragma unroll
      for (int wd = 0; wd < 8; ++wd) {
        const int k0 = hi * 8 + 2 * wd + (wd >= 4 ? 8 : 0);   // A-matrix layout
        a[mt].u[wd] = *(const unsigned*)(&As[r * ASTR + k0]);
      }
    }
#pragma unroll
    for (int nt = 0; nt < 4; ++nt) {
      const int c = wn * 64 + nt * 16 + ln;
#pragma unroll
      for (int wd = 0; wd < 8; ++wd) {
        const int k0 = hi * 8 + 2 * wd + (wd >= 4 ? 8 : 0);   // B-matrix layout
        b[nt].u[wd] = *(const unsigned*)(&Bs[c * ASTR + k0]);
      }
    }
#pragma unroll
    for (int mt = 0; mt < 2; ++mt)
#pragma unroll
      for (int nt = 0; nt < 4; ++nt)
        acc[mt][nt] = WMMA_BF16(a[mt].v, b[nt].v, acc[mt][nt]);
  }

  // Epilogue: bias + scatter into Q / K / V(transposed) layouts
#pragma unroll
  for (int mt = 0; mt < 2; ++mt) {
#pragma unroll
    for (int nt = 0; nt < 4; ++nt) {
      const int o = n0 + wn * 64 + nt * 16 + ln;
      const float bv = bias[o];
#pragma unroll
      for (int v = 0; v < 8; ++v) {
        const int t = m0 + wm * 32 + mt * 16 + v + hi * 8;   // C-layout row
        const float val = acc[mt][nt][v] + bv;
        const int s = t >> 3, bb = t & 7;
        if (o < 1024) {
          const int h = o >> 6, hd = o & 63;
          qb[(((size_t)(bb * NHEAD + h)) * S_TOT + s) * HDIM + hd] = f2bf(val);
        } else if (o < 2048) {
          const int oo = o - 1024, h = oo >> 6, hd = oo & 63;
          if (t < NKV)
            kb[(((size_t)(bb * NHEAD + h)) * SEP + s) * HDIM + hd] = f2bf(val);
        } else {
          const int oo = o - 2048, h = oo >> 6, hd = oo & 63;
          if (t < NKV)
            vb[(((size_t)(bb * NHEAD + h)) * HDIM + hd) * SEP + s] = f2bf(val);
        }
      }
    }
  }
}

// ---------------------------------------------------------------------------
// Kernel 2: flash-style attention with task-id mask.
// Block = 128 threads (4 waves) handles one (b,h) and 64 query rows
// (16 rows per wave).  Loops over the 1024 train keys in tiles of 64,
// double-buffered through LDS via async global->LDS copies (ASYNCcnt).
// ---------------------------------------------------------------------------
__global__ __launch_bounds__(128) void attn_kernel(
    const unsigned short* __restrict__ qb, const unsigned short* __restrict__ kb,
    const unsigned short* __restrict__ vb, const int* __restrict__ task_ids,
    unsigned short* __restrict__ ctx)   // (token, D) bf16
{
  const int qt = blockIdx.x;           // 0..31 (query tile of 64)
  const int bh = blockIdx.y;           // b*16 + h
  const int b = bh >> 4, h = bh & 15;
  const int tid = threadIdx.x, wv = tid >> 5, lane = tid & 31;
  const int hi = lane >> 4, ln = lane & 15;

  const unsigned short* qbase = qb + (size_t)bh * S_TOT * HDIM;
  const unsigned short* kbase = kb + (size_t)bh * SEP * HDIM;
  const unsigned short* vbase = vb + (size_t)bh * HDIM * SEP;

  __shared__ __align__(16) unsigned short Ks[2][64 * KSTR];
  __shared__ __align__(16) unsigned short Vs[2][64 * KSTR];
  __shared__ __align__(16) unsigned short Ps[4 * 16 * KSTR];
  __shared__ int tk[2][64];

  const int qrow0 = qt * 64 + wv * 16;

  // Q fragments (16 x 64 over hd, split into two K=32 halves) from global
  ABf qa[2];
  {
    const int r = qrow0 + ln;
#pragma unroll
    for (int half = 0; half < 2; ++half)
#pragma unroll
      for (int wd = 0; wd < 8; ++wd) {
        const int k0 = half * 32 + hi * 8 + 2 * wd + (wd >= 4 ? 8 : 0);
        qa[half].u[wd] = *(const unsigned*)(qbase + (size_t)r * HDIM + k0);
      }
  }
  int tq[8];
#pragma unroll
  for (int v = 0; v < 8; ++v) tq[v] = task_ids[qrow0 + v + hi * 8];

  floatx8 zero = {};
  floatx8 O[4];
  for (int i = 0; i < 4; ++i) O[i] = zero;
  float mrow[8], lrow[8];
#pragma unroll
  for (int v = 0; v < 8; ++v) { mrow[v] = -3.0e38f; lrow[v] = 0.0f; }

  const int lrw = tid >> 1;
  const int lcg = (tid & 1) * 32;

  // stage one 64-key tile (K, V, key task-ids) into LDS buffer `bufi`
  auto issue_tile = [&](int kt, int bufi) {
    const unsigned short* gk = kbase + (size_t)(kt * 64 + lrw) * HDIM + lcg;
    const unsigned short* gv = vbase + (size_t)lrw * SEP + kt * 64 + lcg;
    unsigned short* dk = &Ks[bufi][lrw * KSTR + lcg];
    unsigned short* dv = &Vs[bufi][lrw * KSTR + lcg];
#if HAVE_ASYNC_LDS
#pragma unroll
    for (int j = 0; j < 4; ++j) {
      __builtin_amdgcn_global_load_async_to_lds_b128(GPTR(gk + j * 8), LPTR(dk + j * 8), 0, 0);
      __builtin_amdgcn_global_load_async_to_lds_b128(GPTR(gv + j * 8), LPTR(dv + j * 8), 0, 0);
    }
#else
#pragma unroll
    for (int j = 0; j < 4; ++j) {
      *(uint4*)(dk + j * 8) = *(const uint4*)(gk + j * 8);
      *(uint4*)(dv + j * 8) = *(const uint4*)(gv + j * 8);
    }
#endif
    if (tid < 64) tk[bufi][tid] = task_ids[kt * 64 + tid];
  };

  issue_tile(0, 0);

  for (int kt = 0; kt < SEP / 64; ++kt) {
    const int bufi = kt & 1;
    ASYNC_WAIT();            // this wave's async copies for tile kt done
    __syncthreads();         // all waves' copies visible; prior compute done
    if (kt + 1 < SEP / 64) issue_tile(kt + 1, bufi ^ 1);

    // S = Q * K^T  (16 x 64), 4 key sub-tiles x 2 WMMAs
    floatx8 S[4];
#pragma unroll
    for (int ns = 0; ns < 4; ++ns) {
      floatx8 s = zero;
#pragma unroll
      for (int half = 0; half < 2; ++half) {
        ABf bf;
#pragma unroll
        for (int wd = 0; wd < 8; ++wd)
          bf.u[wd] = *(const unsigned*)(&Ks[bufi][(ns * 16 + ln) * KSTR + half * 32 + hi * 8 + 2 * wd + (wd >= 4 ? 8 : 0)]);
        s = WMMA_BF16(qa[half].v, bf.v, s);
      }
      S[ns] = s;
    }
    // scale + task-id mask (masked entries -> exactly -1e9 like reference)
#pragma unroll
    for (int ns = 0; ns < 4; ++ns) {
      const int tkk = tk[bufi][ns * 16 + ln];
#pragma unroll
      for (int v = 0; v < 8; ++v) {
        const float x = S[ns][v] * 0.125f;     // 1/sqrt(64)
        S[ns][v] = (tq[v] != tkk) ? -1.0e9f : x;
      }
    }
    // online softmax: row max across 4 tiles + 16 lanes of the half-wave
    float sc[8];
#pragma unroll
    for (int v = 0; v < 8; ++v) {
      float x = fmaxf(fmaxf(S[0][v], S[1][v]), fmaxf(S[2][v], S[3][v]));
#pragma unroll
      for (int off = 1; off < 16; off <<= 1) x = fmaxf(x, __shfl_xor(x, off, 32));
      const float mn = fmaxf(mrow[v], x);
      sc[v] = __expf(mrow[v] - mn);
      mrow[v] = mn;
    }
    float rs[8];
#pragma unroll
    for (int v = 0; v < 8; ++v) rs[v] = 0.0f;
#pragma unroll
    for (int ns = 0; ns < 4; ++ns)
#pragma unroll
      for (int v = 0; v < 8; ++v) {
        const float p = __expf(S[ns][v] - mrow[v]);
        S[ns][v] = p;
        rs[v] += p;
      }
#pragma unroll
    for (int v = 0; v < 8; ++v) {
      float x = rs[v];
#pragma unroll
      for (int off = 1; off < 16; off <<= 1) x += __shfl_xor(x, off, 32);
      lrow[v] = lrow[v] * sc[v] + x;
    }
#pragma unroll
    for (int hs = 0; hs < 4; ++hs)
#pragma unroll
      for (int v = 0; v < 8; ++v) O[hs][v] *= sc[v];

    // stage P (bf16) to wave-private LDS in row-major, reload in A layout
    // (same-wave LDS ops are in-order: no barrier needed)
#pragma unroll
    for (int ns = 0; ns < 4; ++ns)
#pragma unroll
      for (int v = 0; v < 8; ++v)
        Ps[(wv * 16 + v + hi * 8) * KSTR + ns * 16 + ln] = f2bf(S[ns][v]);

    // O += P (16x64) * V (64x64)
#pragma unroll
    for (int half = 0; half < 2; ++half) {
      ABf pa;
#pragma unroll
      for (int wd = 0; wd < 8; ++wd) {
        const int k0 = half * 32 + hi * 8 + 2 * wd + (wd >= 4 ? 8 : 0);
        pa.u[wd] = *(const unsigned*)(&Ps[(wv * 16 + ln) * KSTR + k0]);
      }
#pragma unroll
      for (int hs = 0; hs < 4; ++hs) {
        ABf bf;
#pragma unroll
        for (int wd = 0; wd < 8; ++wd)
          bf.u[wd] = *(const unsigned*)(&Vs[bufi][(hs * 16 + ln) * KSTR + half * 32 + hi * 8 + 2 * wd + (wd >= 4 ? 8 : 0)]);
        O[hs] = WMMA_BF16(pa.v, bf.v, O[hs]);
      }
    }
  }

  // normalize and write ctx (token-major, bf16)
#pragma unroll
  for (int hs = 0; hs < 4; ++hs)
#pragma unroll
    for (int v = 0; v < 8; ++v) {
      const int srow = qrow0 + v + hi * 8;
      const int t = srow * BATCH + b;
      const int oc = h * HDIM + hs * 16 + ln;
      ctx[(size_t)t * DMODEL + oc] = f2bf(O[hs][v] / lrow[v]);
    }
}

// ---------------------------------------------------------------------------
// Kernel 3: out projection + bias + residual  ->  d_out (f32, pre-LN)
// A tile (ctx, already bf16) staged via async global->LDS copy.
// ---------------------------------------------------------------------------
__global__ __launch_bounds__(256) void out_gemm_kernel(
    const unsigned short* __restrict__ ctx, const float* __restrict__ w,
    const float* __restrict__ bias, const float* __restrict__ src,
    float* __restrict__ out)
{
  const int n0 = blockIdx.x * 128;
  const int m0 = blockIdx.y * 128;

  __shared__ __align__(16) unsigned short As[128 * ASTR];
  __shared__ __align__(16) unsigned short Bs[128 * ASTR];

  const int tid = threadIdx.x;
  const int wv = tid >> 5, lane = tid & 31, hi = lane >> 4, ln = lane & 15;
  const int wm = wv >> 1, wn = wv & 1;

  floatx8 zero = {};
  floatx8 acc[2][4];
  for (int i = 0; i < 2; ++i)
    for (int j = 0; j < 4; ++j) acc[i][j] = zero;

  const int lrow = tid >> 1;
  const int lcg  = (tid & 1) * 16;

  for (int kt = 0; kt < DMODEL; kt += 32) {
    __syncthreads();
    {
      const unsigned short* p = ctx + (size_t)(m0 + lrow) * DMODEL + kt + lcg;
#if HAVE_ASYNC_LDS
      __builtin_amdgcn_global_load_async_to_lds_b128(GPTR(p),     LPTR(&As[lrow * ASTR + lcg]),     0, 0);
      __builtin_amdgcn_global_load_async_to_lds_b128(GPTR(p + 8), LPTR(&As[lrow * ASTR + lcg + 8]), 0, 0);
#else
      *(uint4*)(&As[lrow * ASTR + lcg    ]) = *(const uint4*)(p);
      *(uint4*)(&As[lrow * ASTR + lcg + 8]) = *(const uint4*)(p + 8);
#endif
      const float* q = w + (size_t)(n0 + lrow) * DMODEL + kt + lcg;
#pragma unroll
      for (int j = 0; j < 4; ++j) {
        float4 g = *(const float4*)(q + j * 4);
        *(unsigned*)(&Bs[lrow * ASTR + lcg + j * 4    ]) = pack2bf(g.x, g.y);
        *(unsigned*)(&Bs[lrow * ASTR + lcg + j * 4 + 2]) = pack2bf(g.z, g.w);
      }
      if (kt + 32 < DMODEL) __builtin_prefetch(q + 32, 0, 0);
    }
    ASYNC_WAIT();
    __syncthreads();

    ABf a[2], b[4];
#pragma unroll
    for (int mt = 0; mt < 2; ++mt) {
      const int r = wm * 32 + mt * 16 + ln;
#pragma unroll
      for (int wd = 0; wd < 8; ++wd) {
        const int k0 = hi * 8 + 2 * wd + (wd >= 4 ? 8 : 0);
        a[mt].u[wd] = *(const unsigned*)(&As[r * ASTR + k0]);
      }
    }
#pragma unroll
    for (int nt = 0; nt < 4; ++nt) {
      const int c = wn * 64 + nt * 16 + ln;
#pragma unroll
      for (int wd = 0; wd < 8; ++wd) {
        const int k0 = hi * 8 + 2 * wd + (wd >= 4 ? 8 : 0);
        b[nt].u[wd] = *(const unsigned*)(&Bs[c * ASTR + k0]);
      }
    }
#pragma unroll
    for (int mt = 0; mt < 2; ++mt)
#pragma unroll
      for (int nt = 0; nt < 4; ++nt)
        acc[mt][nt] = WMMA_BF16(a[mt].v, b[nt].v, acc[mt][nt]);
  }

#pragma unroll
  for (int mt = 0; mt < 2; ++mt) {
#pragma unroll
    for (int nt = 0; nt < 4; ++nt) {
      const int o = n0 + wn * 64 + nt * 16 + ln;
      const float bv = bias[o];
#pragma unroll
      for (int v = 0; v < 8; ++v) {
        const int t = m0 + wm * 32 + mt * 16 + v + hi * 8;
        out[(size_t)t * DMODEL + o] =
            acc[mt][nt][v] + bv + src[(size_t)t * DMODEL + o];
      }
    }
  }
}

// ---------------------------------------------------------------------------
// Kernel 4: in-place LayerNorm per token row (1024 elems, 256 threads x 4)
// ---------------------------------------------------------------------------
__global__ __launch_bounds__(256) void ln_kernel(
    float* __restrict__ out, const float* __restrict__ g, const float* __restrict__ be)
{
  const int row = blockIdx.x;
  const int tid = threadIdx.x;
  __shared__ float sbuf[256];

  float x[4];
#pragma unroll
  for (int j = 0; j < 4; ++j) x[j] = out[(size_t)row * DMODEL + tid + j * 256];

  float s = x[0] + x[1] + x[2] + x[3];
  sbuf[tid] = s;
  __syncthreads();
  for (int off = 128; off > 0; off >>= 1) {
    if (tid < off) sbuf[tid] += sbuf[tid + off];
    __syncthreads();
  }
  const float mean = sbuf[0] * (1.0f / DMODEL);
  __syncthreads();

  float vs = 0.0f;
#pragma unroll
  for (int j = 0; j < 4; ++j) { const float d = x[j] - mean; vs += d * d; }
  sbuf[tid] = vs;
  __syncthreads();
  for (int off = 128; off > 0; off >>= 1) {
    if (tid < off) sbuf[tid] += sbuf[tid + off];
    __syncthreads();
  }
  const float rstd = rsqrtf(sbuf[0] * (1.0f / DMODEL) + 1e-5f);

#pragma unroll
  for (int j = 0; j < 4; ++j) {
    const int c = tid + j * 256;
    out[(size_t)row * DMODEL + c] = (x[j] - mean) * rstd * g[c] + be[c];
  }
}

// ---------------------------------------------------------------------------
extern "C" void kernel_launch(void* const* d_in, const int* in_sizes, int n_in,
                              void* d_out, int out_size, void* d_ws, size_t ws_size,
                              hipStream_t stream)
{
  const float* src       = (const float*)d_in[0];
  const int*   task_ids  = (const int*)d_in[1];
  // d_in[2] = single_eval_position (fixed at 1024)
  const float* in_proj_w = (const float*)d_in[3];
  const float* in_proj_b = (const float*)d_in[4];
  const float* out_w     = (const float*)d_in[5];
  const float* out_b     = (const float*)d_in[6];
  const float* norm_w    = (const float*)d_in[7];
  const float* norm_b    = (const float*)d_in[8];
  float* out = (float*)d_out;

  char* ws = (char*)d_ws;
  unsigned short* qb  = (unsigned short*)(ws);                 // 33,554,432 B
  unsigned short* kb  = (unsigned short*)(ws + 33554432);      // 16,777,216 B
  unsigned short* vb  = (unsigned short*)(ws + 50331648);      // 16,777,216 B
  unsigned short* ctx = (unsigned short*)(ws + 67108864);      // 33,554,432 B
  (void)in_sizes; (void)n_in; (void)out_size; (void)ws_size;

  dim3 g1(24, 128);
  qkv_gemm_kernel<<<g1, 256, 0, stream>>>(src, in_proj_w, in_proj_b, qb, kb, vb);

  dim3 g2(32, 128);
  attn_kernel<<<g2, 128, 0, stream>>>(qb, kb, vb, task_ids, ctx);

  dim3 g3(8, 128);
  out_gemm_kernel<<<g3, 256, 0, stream>>>(ctx, out_w, out_b, src, out);

  ln_kernel<<<NTOK, 256, 0, stream>>>(out, norm_w, norm_b);
}
